// SAModule_10917806866864
// MI455X (gfx1250) — matrix operand order repaired and checked by reading, not compiled, not run on hardware
//
#include <hip/hip_runtime.h>

// ---------------------------------------------------------------------------
// PointNet++ SA module for MI455X (gfx1250, wave32, WMMA)
//   B=8, N=4096, C=64, M=2048, K=64, HID1=HID2=64, OUT=128, R^2=0.04
// Pipeline: fps (LDS-cached, shuffle argmax) -> weight transpose/f16 ->
//           ball query (LDS-cached cloud, top-K) -> WMMA MLP+pool
//           (8 centroids per block to amortize weight staging)
// ---------------------------------------------------------------------------

#define BB    8
#define NN    4096
#define CC    64
#define MM    2048
#define KK    64
#define OUTC  128
#define R2    0.04f
#define CPB   8      // centroids per block in the MLP kernel

typedef __attribute__((ext_vector_type(16))) _Float16 v16h;
typedef __attribute__((ext_vector_type(8)))  _Float16 v8h;
typedef __attribute__((ext_vector_type(4)))  _Float16 v4h;
typedef __attribute__((ext_vector_type(8)))  float    v8f;

union frag16 { v16h v; v8h h[2]; };

// ---------------------------------------------------------------------------
// Kernel 1: farthest point sampling. One block per cloud, 1024 threads.
// Cloud cached in LDS (48KB); per-step argmax = in-register 4-way +
// 5-step wave shuffle + 32-entry cross-wave stage (2 barriers per step).
// Tie-break: lowest index (matches jnp.argmax).
// ---------------------------------------------------------------------------
__global__ __launch_bounds__(1024) void fps_kernel(const float* __restrict__ pos,
                                                   int* __restrict__ idxArr) {
    __shared__ float spx[NN], spy[NN], spz[NN];
    __shared__ float redV[32];
    __shared__ int   redI[32];
    __shared__ float lastP[3];
    const int b = blockIdx.x, t = threadIdx.x;
    const int lane = t & 31, wid = t >> 5;
    const float* pb = pos + (size_t)b * NN * 3;
    for (int p = t; p < NN; p += 1024) {
        spx[p] = pb[p*3+0]; spy[p] = pb[p*3+1]; spz[p] = pb[p*3+2];
    }
    if (t == 0) {
        idxArr[b * MM] = 0;
        lastP[0] = pb[0]; lastP[1] = pb[1]; lastP[2] = pb[2];
    }
    float mind[4] = {3.4e38f, 3.4e38f, 3.4e38f, 3.4e38f};
    __syncthreads();
    for (int s = 1; s < MM; s++) {
        const float lx = lastP[0], ly = lastP[1], lz = lastP[2];
        float bv = -1.0f; int bi = 0x7fffffff;
        #pragma unroll
        for (int i = 0; i < 4; i++) {
            const int p = t + i * 1024;
            const float dx = spx[p] - lx, dy = spy[p] - ly, dz = spz[p] - lz;
            const float d = dx*dx + dy*dy + dz*dz;
            mind[i] = fminf(mind[i], d);
            if (mind[i] > bv || (mind[i] == bv && p < bi)) { bv = mind[i]; bi = p; }
        }
        // wave32 argmax reduction
        #pragma unroll
        for (int off = 16; off > 0; off >>= 1) {
            const float ov = __shfl_down(bv, off, 32);
            const int   oi = __shfl_down(bi, off, 32);
            if (ov > bv || (ov == bv && oi < bi)) { bv = ov; bi = oi; }
        }
        if (lane == 0) { redV[wid] = bv; redI[wid] = bi; }
        __syncthreads();
        if (wid == 0) {
            bv = redV[lane]; bi = redI[lane];
            #pragma unroll
            for (int off = 16; off > 0; off >>= 1) {
                const float ov = __shfl_down(bv, off, 32);
                const int   oi = __shfl_down(bi, off, 32);
                if (ov > bv || (ov == bv && oi < bi)) { bv = ov; bi = oi; }
            }
            if (lane == 0) {
                idxArr[b * MM + s] = bi;
                lastP[0] = spx[bi]; lastP[1] = spy[bi]; lastP[2] = spz[bi];
            }
        }
        __syncthreads();
    }
}

// ---------------------------------------------------------------------------
// Kernel 2: weight prep. Transpose W's to [out_col][k] f16 so WMMA B-fragments
// load as contiguous K-runs. W1t padded to k=96 (cols 67..95 zero).
// ---------------------------------------------------------------------------
__global__ void prep_weights(const float* __restrict__ W1, const float* __restrict__ W2,
                             const float* __restrict__ W3, _Float16* __restrict__ W1t,
                             _Float16* __restrict__ W2t, _Float16* __restrict__ W3t) {
    const int stride = gridDim.x * blockDim.x;
    const int tid = blockIdx.x * blockDim.x + threadIdx.x;
    for (int i = tid; i < 64 * 96; i += stride) {
        const int n = i / 96, k = i % 96;
        W1t[i] = (k < 67) ? (_Float16)W1[k * 64 + n] : (_Float16)0.0f;
    }
    for (int i = tid; i < 64 * 64; i += stride) {
        const int n = i / 64, k = i % 64;
        W2t[i] = (_Float16)W2[k * 64 + n];
    }
    for (int i = tid; i < 128 * 64; i += stride) {
        const int n = i / 64, k = i % 64;
        W3t[i] = (_Float16)W3[k * 128 + n];
    }
}

// ---------------------------------------------------------------------------
// Kernel 3: radius ball query, K-closest within R (== top_k(-d2) incl. tie
// order). Cloud cached in LDS once per block (8 centroids share it). One wave
// per centroid; lanes compute 32 d2's, lane 0 does a stable sorted insertion
// into a per-wave LDS list capped at K.
// ---------------------------------------------------------------------------
__global__ __launch_bounds__(256) void ball_kernel(const float* __restrict__ pos,
                                                   const int* __restrict__ idxArr,
                                                   int* __restrict__ nbrArr,
                                                   int* __restrict__ cntArr,
                                                   float* __restrict__ posOut,
                                                   float* __restrict__ batchOut) {
    __shared__ float qx[NN], qy[NN], qz[NN];
    __shared__ float ld[8][KK];
    __shared__ int   li[8][KK];
    __shared__ float sd2[8][32];
    __shared__ int   lcnt[8];
    const int tid = threadIdx.x, lane = tid & 31, wid = tid >> 5;
    const int g = blockIdx.x * 8 + wid;
    const int b = g >> 11;  // M = 2048; all 8 centroids in a block share cloud b
    const float* pb = pos + (size_t)b * NN * 3;
    for (int p = tid; p < NN; p += 256) {
        qx[p] = pb[p*3+0]; qy[p] = pb[p*3+1]; qz[p] = pb[p*3+2];
    }
    __syncthreads();
    const int ci = idxArr[g];
    const float cx = qx[ci], cy = qy[ci], cz = qz[ci];
    int c = 0;
    for (int base = 0; base < NN; base += 32) {
        const int p = base + lane;
        const float dx = qx[p] - cx, dy = qy[p] - cy, dz = qz[p] - cz;
        const float d2v = dx*dx + dy*dy + dz*dz;
        sd2[wid][lane] = d2v;
        unsigned msk = (unsigned)__ballot(d2v <= R2);
        if (lane == 0) {
            while (msk) {
                const int j = __builtin_ctz(msk);
                msk &= msk - 1;
                const float dj = sd2[wid][j];
                const int pj = base + j;
                if (c == KK && dj >= ld[wid][KK-1]) continue;
                int ip = (c < KK) ? c : (KK - 1);
                while (ip > 0 && ld[wid][ip-1] > dj) {
                    ld[wid][ip] = ld[wid][ip-1];
                    li[wid][ip] = li[wid][ip-1];
                    ip--;
                }
                ld[wid][ip] = dj; li[wid][ip] = pj;
                if (c < KK) c++;
            }
        }
    }
    if (lane == 0) {
        lcnt[wid] = c;
        cntArr[g] = c;
        posOut[g*3+0] = cx; posOut[g*3+1] = cy; posOut[g*3+2] = cz;
        batchOut[g] = (float)b;
    }
    const int cc = lcnt[wid];
    for (int s = lane; s < KK; s += 32)
        nbrArr[(size_t)g * KK + s] = (s < cc) ? li[wid][s] : 0;
}

// ---------------------------------------------------------------------------
// WMMA fragment loaders (per ISA 7.12.2, wave32, 16-bit 16x16x32):
//   A (16x32): lane L<16 row L, halves 0..7 = K kb..kb+7, 8..15 = K kb+16..+23
//              with kb = 0 (L<16) or 8 (L>=16)
//   B (32x16): lane L holds column L%16, same K order as A:
//              halves 0..7 = K kb..kb+7, 8..15 = K kb+16..+23
// ---------------------------------------------------------------------------
static __device__ __forceinline__ v16h load_frag_A(const _Float16* base, int stride,
                                                   int row0, int k0, int lane) {
    const int r  = row0 + (lane & 15);
    const int kb = k0 + ((lane >> 4) << 3);
    frag16 f;
    f.h[0] = *(const v8h*)(base + r * stride + kb);
    f.h[1] = *(const v8h*)(base + r * stride + kb + 16);
    return f.v;
}

static __device__ __forceinline__ v16h load_frag_B(const _Float16* base, int stride,
                                                   int col0, int k0, int lane) {
    const int n  = col0 + (lane & 15);
    const int kb = k0 + ((lane >> 4) << 3);
    frag16 f;
    f.h[0] = *(const v8h*)(base + n * stride + kb);
    f.h[1] = *(const v8h*)(base + n * stride + kb + 16);
    return f.v;
}

// C/D layout: lane L -> col L%16; element r -> row 8*(L/16)+r within the tile.
static __device__ __forceinline__ void store_tile_relu(_Float16* dst, int stride,
                                                       int row0, int col0, int lane,
                                                       v8f acc, float bias) {
    const int ccol  = col0 + (lane & 15);
    const int rbase = row0 + ((lane >> 4) << 3);
    #pragma unroll
    for (int r = 0; r < 8; r++) {
        float v = acc[r] + bias;
        v = v > 0.0f ? v : 0.0f;
        dst[(rbase + r) * stride + ccol] = (_Float16)v;
    }
}

#define WMMA_F16(acc, a, bf) \
    acc = __builtin_amdgcn_wmma_f32_16x16x32_f16(false, (a), false, (bf), (short)0, (acc), false, false)

// ---------------------------------------------------------------------------
// Kernel 4: grouped MLP + masked max pool. One block (4 waves) handles CPB=8
// centroids, staging the f16 weights + biases in LDS once. Per centroid:
// gather -> f16 LDS -> 3 WMMA layers (wave w owns rows 16w..16w+15, so no
// inter-layer barriers) -> ReLU -> masked max via LDS u32 atomicMax
// (valid since ReLU output >= 0; count==0 row stays 0 as in the reference).
// ---------------------------------------------------------------------------
__global__ __launch_bounds__(128) void mlp_kernel(const float* __restrict__ x,
                                                  const float* __restrict__ pos,
                                                  const int* __restrict__ idxArr,
                                                  const int* __restrict__ nbrArr,
                                                  const int* __restrict__ cntArr,
                                                  const _Float16* __restrict__ W1t,
                                                  const _Float16* __restrict__ W2t,
                                                  const _Float16* __restrict__ W3t,
                                                  const float* __restrict__ b1,
                                                  const float* __restrict__ b2,
                                                  const float* __restrict__ b3,
                                                  float* __restrict__ out) {
    __shared__ __align__(16) _Float16 sH0[64 * 96];   // gathered [x_j, rel, pad]
    __shared__ __align__(16) _Float16 sW1[64 * 96];   // [n][k]
    __shared__ __align__(16) _Float16 sW2[64 * 64];
    __shared__ __align__(16) _Float16 sW3[128 * 64];
    __shared__ __align__(16) _Float16 sH [64 * 64];   // H1 then H2 (per-wave rows)
    __shared__ float sB1[64], sB2[64], sB3[OUTC];
    __shared__ unsigned int sOut[OUTC];

    const int tid = threadIdx.x, lane = tid & 31, wid = tid >> 5;
    const int row0 = wid * 16;

    // --- stage weights + biases in LDS once per block (16B chunks)
    {
        const uint4* s = (const uint4*)W1t; uint4* d = (uint4*)sW1;
        for (int i = tid; i < 768; i += 128) d[i] = s[i];
        s = (const uint4*)W2t; d = (uint4*)sW2;
        for (int i = tid; i < 512; i += 128) d[i] = s[i];
        s = (const uint4*)W3t; d = (uint4*)sW3;
        for (int i = tid; i < 1024; i += 128) d[i] = s[i];
        if (tid < 64) { sB1[tid] = b1[tid]; sB2[tid] = b2[tid]; }
        sB3[tid] = b3[tid];
        sOut[tid] = 0u;  // bits of 0.0f
    }

    for (int cc = 0; cc < CPB; cc++) {
        const int g = blockIdx.x * CPB + cc;
        const int b = g >> 11;  // M = 2048
        const int* nbr = nbrArr + (size_t)g * KK;

        const int ci = idxArr[g];
        const float* cp = pos + ((size_t)b * NN + ci) * 3;
        const float cx = cp[0], cy = cp[1], cz = cp[2];

        // --- gather features: 2 threads per neighbor row, 32 channels each
        {
            const int row = tid >> 1, half = tid & 1;
            const int j = nbr[row];
            const float* src = x + ((size_t)b * NN + j) * CC + half * 32;
            _Float16* dst = sH0 + row * 96 + half * 32;
            #pragma unroll
            for (int i = 0; i < 32; i += 4) {
                const float4 f = *(const float4*)(src + i);
                v4h h = {(_Float16)f.x, (_Float16)f.y, (_Float16)f.z, (_Float16)f.w};
                *(v4h*)(dst + i) = h;
            }
        }
        // --- relative position + zero padding
        if (tid < 64) {
            const int row = tid;
            const int j = nbr[row];
            const float* pj = pos + ((size_t)b * NN + j) * 3;
            sH0[row * 96 + 64] = (_Float16)(pj[0] - cx);
            sH0[row * 96 + 65] = (_Float16)(pj[1] - cy);
            sH0[row * 96 + 66] = (_Float16)(pj[2] - cz);
            for (int k = 67; k < 96; k++) sH0[row * 96 + k] = (_Float16)0.0f;
        }
        __syncthreads();

        // --- layer 1: [16x96] @ [96x64]
        const v16h a0 = load_frag_A(sH0, 96, row0, 0,  lane);
        const v16h a1 = load_frag_A(sH0, 96, row0, 32, lane);
        const v16h a2 = load_frag_A(sH0, 96, row0, 64, lane);
        #pragma unroll
        for (int t = 0; t < 4; t++) {
            v8f acc = {};
            WMMA_F16(acc, a0, load_frag_B(sW1, 96, t * 16, 0,  lane));
            WMMA_F16(acc, a1, load_frag_B(sW1, 96, t * 16, 32, lane));
            WMMA_F16(acc, a2, load_frag_B(sW1, 96, t * 16, 64, lane));
            store_tile_relu(sH, 64, row0, t * 16, lane, acc, sB1[t * 16 + (lane & 15)]);
        }

        // --- layer 2: [16x64] @ [64x64]  (wave reads only its own rows)
        const v16h c0 = load_frag_A(sH, 64, row0, 0,  lane);
        const v16h c1 = load_frag_A(sH, 64, row0, 32, lane);
        #pragma unroll
        for (int t = 0; t < 4; t++) {
            v8f acc = {};
            WMMA_F16(acc, c0, load_frag_B(sW2, 64, t * 16, 0,  lane));
            WMMA_F16(acc, c1, load_frag_B(sW2, 64, t * 16, 32, lane));
            store_tile_relu(sH, 64, row0, t * 16, lane, acc, sB2[t * 16 + (lane & 15)]);
        }

        // --- layer 3: [16x64] @ [64x128] + masked row max
        const v16h d0 = load_frag_A(sH, 64, row0, 0,  lane);
        const v16h d1 = load_frag_A(sH, 64, row0, 32, lane);
        const int cnt = cntArr[g];
        const int rbase = row0 + ((lane >> 4) << 3);
        #pragma unroll
        for (int t = 0; t < 8; t++) {
            v8f acc = {};
            WMMA_F16(acc, d0, load_frag_B(sW3, 64, t * 16, 0,  lane));
            WMMA_F16(acc, d1, load_frag_B(sW3, 64, t * 16, 32, lane));
            const int ccol = t * 16 + (lane & 15);
            const float bias = sB3[ccol];
            float pmax = 0.0f;
            #pragma unroll
            for (int r = 0; r < 8; r++) {
                float v = acc[r] + bias;
                v = v > 0.0f ? v : 0.0f;
                if (rbase + r < cnt) pmax = fmaxf(pmax, v);
            }
            atomicMax(&sOut[ccol], __float_as_uint(pmax));
        }
        __syncthreads();
        // each thread reads and resets its own slot: no cross-thread race
        out[(size_t)g * OUTC + tid] = __uint_as_float(sOut[tid]);
        sOut[tid] = 0u;
    }
}

// ---------------------------------------------------------------------------
// Host launcher
// ---------------------------------------------------------------------------
extern "C" void kernel_launch(void* const* d_in, const int* in_sizes, int n_in,
                              void* d_out, int out_size, void* d_ws, size_t ws_size,
                              hipStream_t stream) {
    (void)in_sizes; (void)n_in; (void)out_size; (void)ws_size;
    const float* x   = (const float*)d_in[0];
    const float* pos = (const float*)d_in[1];
    // d_in[2] = batch (unused: batch_s == cloud index)
    const float* W1  = (const float*)d_in[3];
    const float* b1  = (const float*)d_in[4];
    const float* W2  = (const float*)d_in[5];
    const float* b2  = (const float*)d_in[6];
    const float* W3  = (const float*)d_in[7];
    const float* b3  = (const float*)d_in[8];

    float* out      = (float*)d_out;
    float* posOut   = out + (size_t)BB * MM * OUTC;
    float* batchOut = posOut + (size_t)BB * MM * 3;

    char* ws = (char*)d_ws;
    int* idxArr = (int*)ws;                                   // 16384 * 4
    int* nbrArr = (int*)(ws + 65536);                         // 16384 * 64 * 4
    int* cntArr = (int*)(ws + 65536 + 4194304);               // 16384 * 4
    _Float16* W1t = (_Float16*)(ws + 4325376);                // 64*96*2
    _Float16* W2t = (_Float16*)(ws + 4337664);                // 64*64*2
    _Float16* W3t = (_Float16*)(ws + 4345856);                // 128*64*2

    fps_kernel<<<dim3(BB), dim3(1024), 0, stream>>>(pos, idxArr);
    prep_weights<<<dim3(16), dim3(256), 0, stream>>>(W1, W2, W3, W1t, W2t, W3t);
    ball_kernel<<<dim3(BB * MM / 8), dim3(256), 0, stream>>>(pos, idxArr, nbrArr,
                                                             cntArr, posOut, batchOut);
    mlp_kernel<<<dim3(BB * MM / CPB), dim3(128), 0, stream>>>(x, pos, idxArr, nbrArr,
                                                              cntArr, W1t, W2t, W3t,
                                                              b1, b2, b3, out);
}
